// LC_Block_87033217286510
// MI455X (gfx1250) — matrix operand order repaired and checked by reading, not compiled, not run on hardware
//
#include <hip/hip_runtime.h>
#include <hip/hip_bf16.h>
#include <math.h>

// Shapes (from reference): B=16, F1=16, D=2, C=64, T=1000, K=64, C2=32, PW=16
#define Bn 16
#define F1n 16
#define Cn 64
#define Tn 1000
#define TnP 1008        // padded time dim (63 tiles * 16) -> unconditional stores
#define Kn 64
#define C2n 32
#define PWn 16
#define TPn 62          // T//16
#define TPAD 1072       // 31 (front pad) + 1000 + 41 tail (covers max idx 1070)
#define EPSf 1e-5f

typedef __attribute__((ext_vector_type(16))) _Float16 v16h;
typedef __attribute__((ext_vector_type(8)))  float    v8f;
// 4-float vector with only dword alignment guaranteed.
typedef __attribute__((ext_vector_type(4), aligned(4))) float vf4u;
// Fully aligned 4-float vector (for known 16B-aligned addresses).
typedef __attribute__((ext_vector_type(4))) float vf4;

__device__ __forceinline__ float eluf(float x)  { return x > 0.f ? x : (__expf(x) - 1.f); }
__device__ __forceinline__ float sigmf(float x) { return 1.f / (1.f + __expf(-x)); }

// ---------------------------------------------------------------------------
// Kernel 0a: zero-pad x into x_pad [B*C][TPAD]: 31 zeros, 1000 data, zero tail.
// ---------------------------------------------------------------------------
__global__ void k0_pad(const float* __restrict__ x, float* __restrict__ xpad)
{
    const int idx = blockIdx.x * blockDim.x + threadIdx.x;   // < B*C*TPAD
    const int tp = idx % TPAD;
    const int bc = idx / TPAD;
    const int ts = tp - 31;
    xpad[idx] = (ts >= 0 && ts < Tn) ? x[bc * Tn + ts] : 0.f;
}

// ---------------------------------------------------------------------------
// Kernel 0b: fold conv1 bias + BN1 into per-channel scale/offset:
//   bn(acc + cb) = acc*inv + (cb*inv + b1 - m1*inv)
// ---------------------------------------------------------------------------
__global__ void k0_prep(const float* __restrict__ cb,
                        const float* __restrict__ g1, const float* __restrict__ b1,
                        const float* __restrict__ m1, const float* __restrict__ v1,
                        float* __restrict__ bnA, float* __restrict__ bnC)
{
    const int f = threadIdx.x;
    if (f < F1n) {
        float inv = g1[f] * rsqrtf(v1[f] + EPSf);
        bnA[f] = inv;
        bnC[f] = cb[f] * inv + b1[f] - m1[f] * inv;
    }
}

// ---------------------------------------------------------------------------
// Kernel 1: temporal conv1 (K=64, SAME) + folded bias/BN1 + ELU
//           -> h [B,F1,C,TnP] (time-padded); fused xgT = mean over F1,
//           stored TRANSPOSED as [b][t][c] with t padded to TnP.
// One wave computes a 16(F1) x 16(t) tile for one (b,c) via 2x WMMA f16 (K=64).
// All loads & stores unconditional (padded buffers).
// ---------------------------------------------------------------------------
__global__ void k1_conv1_bn_elu(const float* __restrict__ xpad,
                                const float* __restrict__ w,    // conv1_w [F1,K]
                                const float* __restrict__ bnA,  // folded scale [F1]
                                const float* __restrict__ bnC,  // folded offset [F1]
                                float* __restrict__ h, float* __restrict__ xgT)
{
    const int lane = threadIdx.x & 31;
    const int wid  = (blockIdx.x * blockDim.x + threadIdx.x) >> 5;
    const int nTiles = Bn * Cn * 63;           // 63 = ceil(1000/16)
    if (wid >= nTiles) return;                 // wave-uniform

    const int tt = wid % 63;
    const int bc = wid / 63;
    const int c  = bc % Cn;
    const int b  = bc / Cn;
    const int t0 = tt * 16;

    const int  row = lane & 15;
    const bool hi  = lane >= 16;
    const float* xp = xpad + bc * TPAD;        // padded row; index = t + 31

    v8f acc = {};
#pragma unroll
    for (int q = 0; q < 2; ++q) {
        const int kb = q * 32;
        // A: weights, 16x32 f16 tile. lanes 0-15: M=lane, K=0..15;
        //                             lanes16-31: M=lane-16, K=16..31
        v16h A;
#pragma unroll
        for (int i = 0; i < 16; ++i) {
            int kl = (hi ? 16 : 0) + i;
            A[i] = (_Float16)w[row * Kn + kb + kl];
        }
        // B: im2col patches, 32x16. lanes 0-15: N=lane, K=kb+0..15;
        //                           lanes16-31: N=lane-16, K=kb+16..31
        v16h Bm;
        const int base = t0 + row + kb + (hi ? 16 : 0);
        const vf4u* p = (const vf4u*)(xp + base);
        vf4u f0 = p[0], f1 = p[1], f2 = p[2], f3 = p[3];
#pragma unroll
        for (int i = 0; i < 4; ++i) {
            Bm[i]      = (_Float16)f0[i];
            Bm[4 + i]  = (_Float16)f1[i];
            Bm[8 + i]  = (_Float16)f2[i];
            Bm[12 + i] = (_Float16)f3[i];
        }
        acc = __builtin_amdgcn_wmma_f32_16x16x32_f16(false, A, false, Bm,
                                                     (short)0, acc, false, false);
    }

    // Epilogue: D layout — lanes 0-15: N=lane, VGPR r -> M=r; lanes16-31: M=r+8
    // Folded BN params: 8 consecutive floats starting at hi*8 (32B aligned).
    const vf4* pA = (const vf4*)(bnA + (hi ? 8 : 0));
    const vf4* pC = (const vf4*)(bnC + (hi ? 8 : 0));
    vf4 aLo = pA[0], aHi = pA[1];
    vf4 cLo = pC[0], cHi = pC[1];

    const int t = t0 + row;                    // < 1008 always
    float* hp = h + ((size_t)(b * F1n + (hi ? 8 : 0)) * Cn + c) * TnP + t;
    float fsum = 0.f;
#pragma unroll
    for (int r = 0; r < 8; ++r) {
        float scl = (r < 4) ? aLo[r & 3] : aHi[r & 3];
        float ofs = (r < 4) ? cLo[r & 3] : cHi[r & 3];
        float val = eluf(fmaf(acc[r], scl, ofs));
        fsum += val;
        hp[(size_t)r * Cn * TnP] = val;        // unconditional (padded h)
    }
    fsum += __shfl_xor(fsum, 16, 32);          // combine complementary M halves
    if (!hi) xgT[(b * TnP + t) * Cn + c] = fsum * (1.f / F1n);
}

// ---------------------------------------------------------------------------
// Kernel 2: GCN propagate  ax[b,c,t] = sum_j a_hat[c,j] * xgT[b,t,j]  via WMMA
// One wave: 16(c) x 16(t) tile for one b; K = C = 64 -> 2 WMMAs.
// xgT is [b][t][c] (t padded), so each lane's 16 B elements are 16 consecutive
// floats at a 64B-aligned address -> 4x global_load_b128. Rows t>=1000 are
// clamped on read (they hold garbage); ax stores are unconditional (padded).
// ---------------------------------------------------------------------------
__global__ void k2_gcn(const float* __restrict__ ahat,
                       const float* __restrict__ xgT,
                       float* __restrict__ ax)
{
    const int lane = threadIdx.x & 31;
    const int wid  = (blockIdx.x * blockDim.x + threadIdx.x) >> 5;
    const int nTiles = Bn * 4 * 63;
    if (wid >= nTiles) return;

    const int tt = wid % 63;
    const int r2 = wid / 63;
    const int ct = r2 % 4;
    const int b  = r2 / 4;
    const int c0 = ct * 16;
    const int t0 = tt * 16;

    const int  row = lane & 15;
    const bool hi  = lane >= 16;
    const int  t   = t0 + row;
    const int  tc  = min(t, Tn - 1);           // branchless clamp

    v8f acc = {};
#pragma unroll
    for (int q = 0; q < 2; ++q) {
        const int kb = q * 32;
        v16h A;
#pragma unroll
        for (int i = 0; i < 16; ++i) {
            int kl = (hi ? 16 : 0) + i;
            A[i] = (_Float16)ahat[(c0 + row) * Cn + kb + kl];
        }
        v16h Bm;
        const int kofs = kb + (hi ? 16 : 0);   // multiple of 16 -> 64B aligned
        const vf4* p = (const vf4*)(xgT + (size_t)(b * TnP + tc) * Cn + kofs);
        vf4 f0 = p[0], f1 = p[1], f2 = p[2], f3 = p[3];
#pragma unroll
        for (int i = 0; i < 4; ++i) {
            Bm[i]      = (_Float16)f0[i];
            Bm[4 + i]  = (_Float16)f1[i];
            Bm[8 + i]  = (_Float16)f2[i];
            Bm[12 + i] = (_Float16)f3[i];
        }
        acc = __builtin_amdgcn_wmma_f32_16x16x32_f16(false, A, false, Bm,
                                                     (short)0, acc, false, false);
    }
    float* axp = ax + ((size_t)(b * Cn + c0 + (hi ? 8 : 0))) * TnP + t;
#pragma unroll
    for (int r = 0; r < 8; ++r) {
        axp[(size_t)r * TnP] = acc[r];         // unconditional (padded ax)
    }
}

// ---------------------------------------------------------------------------
// Kernel 3: depthwise expansion (groups=F1) with GCN residual folded in,
//           + dw bias + BN2 -> h2 [B,C2,T]; fused channel-attn pooling
//           s[b,j] = mean_t(h2) + max_t(h2).
// One block per (b,j), 256 threads over t.  (h and ax have TnP-padded rows.)
// ---------------------------------------------------------------------------
__global__ void k3_dwexp(const float* __restrict__ h, const float* __restrict__ ax,
                         const float* __restrict__ dw_w, const float* __restrict__ dw_b,
                         const float* __restrict__ gcn_w, const float* __restrict__ gcn_b,
                         const float* __restrict__ g2, const float* __restrict__ b2,
                         const float* __restrict__ m2, const float* __restrict__ v2,
                         float* __restrict__ h2, float* __restrict__ s)
{
    __shared__ float wds[Cn];
    __shared__ float redS[256];
    __shared__ float redM[256];

    const int j = blockIdx.x & (C2n - 1);
    const int b = blockIdx.x >> 5;
    const int f = j >> 1;
    const int tid = threadIdx.x;

    if (tid < Cn) wds[tid] = dw_w[j * Cn + tid];   // dw_w[(f*D+d)*C + c], j=f*2+d
    __syncthreads();

    float swd = 0.f;
#pragma unroll
    for (int cc = 0; cc < Cn; ++cc) swd += wds[cc];

    const float inv = g2[j] * rsqrtf(v2[j] + EPSf);
    const float off = b2[j] - m2[j] * inv;
    const float gw = gcn_w[f], gb = gcn_b[f], db = dw_b[j];

    float lsum = 0.f, lmax = -INFINITY;
    for (int t = tid; t < Tn; t += 256) {
        const float* hp  = h  + ((size_t)(b * F1n + f) * Cn) * TnP + t;
        const float* axp = ax + ((size_t)b * Cn) * TnP + t;
        float a1 = 0.f, a2 = 0.f;
#pragma unroll 8
        for (int cc = 0; cc < Cn; ++cc) {
            a1 += hp[(size_t)cc * TnP]  * wds[cc];
            a2 += axp[(size_t)cc * TnP] * wds[cc];
        }
        float val = (a1 + gw * a2 + gb * swd + db) * inv + off;
        h2[(b * C2n + j) * Tn + t] = val;
        lsum += val;
        lmax = fmaxf(lmax, val);
    }
    redS[tid] = lsum; redM[tid] = lmax;
    __syncthreads();
    for (int st = 128; st > 0; st >>= 1) {
        if (tid < st) {
            redS[tid] += redS[tid + st];
            redM[tid] = fmaxf(redM[tid], redM[tid + st]);
        }
        __syncthreads();
    }
    if (tid == 0) s[b * C2n + j] = redS[0] * (1.f / Tn) + redM[0];
}

// ---------------------------------------------------------------------------
// Kernel 4: squeeze-excite MLP: att = sigmoid(relu(s@W1^T+b1)@W2^T+b2)
// One block, 512 threads = (b, j).
// ---------------------------------------------------------------------------
__global__ void k4_att(const float* __restrict__ s,
                       const float* __restrict__ w1, const float* __restrict__ bb1,
                       const float* __restrict__ w2, const float* __restrict__ bb2,
                       float* __restrict__ att)
{
    const int tid = threadIdx.x;            // 512
    const int b = tid >> 5, j = tid & 31;
    float h0 = bb1[0], h1 = bb1[1];
#pragma unroll
    for (int jj = 0; jj < C2n; ++jj) {
        float sv = s[b * C2n + jj];
        h0 += sv * w1[jj];                  // ca_w1[0, jj]
        h1 += sv * w1[C2n + jj];            // ca_w1[1, jj]
    }
    h0 = fmaxf(h0, 0.f); h1 = fmaxf(h1, 0.f);
    float p = h0 * w2[j * 2 + 0] + h1 * w2[j * 2 + 1] + bb2[j];
    att[b * C2n + j] = sigmf(p);
}

// ---------------------------------------------------------------------------
// Kernel 5: channel-att apply + spatial attention (3-wide, kh=1 only since H=1)
//           + BN + sigmoid gate + ELU + avgpool(16) -> hp [B,C2,62]
// One block per b.
// ---------------------------------------------------------------------------
__global__ void k5_spatial(const float* __restrict__ h2, const float* __restrict__ att,
                           const float* __restrict__ sa_w,
                           const float* __restrict__ sa_g, const float* __restrict__ sa_b,
                           const float* __restrict__ sa_m, const float* __restrict__ sa_v,
                           float* __restrict__ hp)
{
    __shared__ float attS[C2n];
    __shared__ float scM[Tn];
    __shared__ float scX[Tn];
    const int b = blockIdx.x, tid = threadIdx.x;

    if (tid < C2n) attS[tid] = att[b * C2n + tid];
    __syncthreads();

    for (int t = tid; t < Tn; t += 256) {
        float sm = 0.f, mx = -INFINITY;
#pragma unroll
        for (int c2 = 0; c2 < C2n; ++c2) {
            float v = h2[(b * C2n + c2) * Tn + t] * attS[c2];
            sm += v; mx = fmaxf(mx, v);
        }
        scM[t] = sm * (1.f / C2n);
        scX[t] = mx;
    }
    __syncthreads();

    // sa_w flat (1,2,3,3): mean channel kh=1 -> idx 3..5, max channel -> 12..14
    const float wm0 = sa_w[3],  wm1 = sa_w[4],  wm2 = sa_w[5];
    const float wx0 = sa_w[12], wx1 = sa_w[13], wx2 = sa_w[14];
    const float inv = sa_g[0] * rsqrtf(sa_v[0] + EPSf);
    const float off = sa_b[0] - sa_m[0] * inv;

    for (int o = tid; o < C2n * TPn; o += 256) {
        const int c2 = o / TPn, tp = o % TPn;
        const float a = attS[c2];
        float accp = 0.f;
#pragma unroll
        for (int i = 0; i < PWn; ++i) {
            const int t = tp * PWn + i;                 // t <= 991, t+1 always valid
            float sa = wm1 * scM[t] + wx1 * scX[t]
                     + wm2 * scM[t + 1] + wx2 * scX[t + 1];
            if (t > 0) sa += wm0 * scM[t - 1] + wx0 * scX[t - 1];
            float sg = sigmf(sa * inv + off);
            accp += eluf(h2[(b * C2n + c2) * Tn + t] * a * sg);
        }
        hp[(b * C2n + c2) * TPn + tp] = accp * (1.f / PWn);
    }
}

// ---------------------------------------------------------------------------
// Kernel 6: depthwise separable conv (k=16, SAME pad 7/8) + bias + BN3 + ELU
//           + avgpool(16) -> out [B,C2,3].  One block of 512 threads = (b,c2).
// ---------------------------------------------------------------------------
__global__ void k6_sep(const float* __restrict__ hp,
                       const float* __restrict__ sep_w, const float* __restrict__ sep_b,
                       const float* __restrict__ g3, const float* __restrict__ b3,
                       const float* __restrict__ m3, const float* __restrict__ v3,
                       float* __restrict__ out)
{
    const int tid = threadIdx.x;            // 512
    const int b = tid >> 5, c2 = tid & 31;
    const float* r = hp + (b * C2n + c2) * TPn;
    float wv[PWn];
#pragma unroll
    for (int k = 0; k < PWn; ++k) wv[k] = sep_w[c2 * PWn + k];
    const float inv = g3[c2] * rsqrtf(v3[c2] + EPSf);
    const float off = b3[c2] - m3[c2] * inv;
    const float sb = sep_b[c2];

#pragma unroll
    for (int p = 0; p < 3; ++p) {
        float acc = 0.f;
#pragma unroll
        for (int i = 0; i < PWn; ++i) {
            const int t = p * PWn + i;
            float y = sb;
#pragma unroll
            for (int k = 0; k < PWn; ++k) {
                const int u = t + k - 7;                // pad_low = 7
                if (u >= 0 && u < TPn) y += r[u] * wv[k];
            }
            acc += eluf(y * inv + off);
        }
        out[(b * C2n + c2) * 3 + p] = acc * (1.f / PWn);
    }
}

// ---------------------------------------------------------------------------
extern "C" void kernel_launch(void* const* d_in, const int* in_sizes, int n_in,
                              void* d_out, int out_size, void* d_ws, size_t ws_size,
                              hipStream_t stream) {
    const float* x      = (const float*)d_in[0];
    const float* a_hat  = (const float*)d_in[1];
    const float* conv1w = (const float*)d_in[2];
    const float* conv1b = (const float*)d_in[3];
    const float* g1 = (const float*)d_in[4];
    const float* b1 = (const float*)d_in[5];
    const float* m1 = (const float*)d_in[6];
    const float* v1 = (const float*)d_in[7];
    const float* gcn_w = (const float*)d_in[8];
    const float* gcn_b = (const float*)d_in[9];
    const float* dw_w  = (const float*)d_in[10];
    const float* dw_b  = (const float*)d_in[11];
    const float* g2 = (const float*)d_in[12];
    const float* b2 = (const float*)d_in[13];
    const float* m2 = (const float*)d_in[14];
    const float* v2 = (const float*)d_in[15];
    const float* ca_w1 = (const float*)d_in[16];
    const float* ca_b1 = (const float*)d_in[17];
    const float* ca_w2 = (const float*)d_in[18];
    const float* ca_b2 = (const float*)d_in[19];
    const float* sa_w  = (const float*)d_in[20];
    const float* sa_g  = (const float*)d_in[21];
    const float* sa_b  = (const float*)d_in[22];
    const float* sa_m  = (const float*)d_in[23];
    const float* sa_v  = (const float*)d_in[24];
    const float* sep_w = (const float*)d_in[25];
    const float* sep_b = (const float*)d_in[26];
    const float* g3 = (const float*)d_in[27];
    const float* b3 = (const float*)d_in[28];
    const float* m3 = (const float*)d_in[29];
    const float* v3 = (const float*)d_in[30];
    float* out = (float*)d_out;

    // Workspace layout (floats). Total ~81 MB.
    float* ws   = (float*)d_ws;
    float* h    = ws;                                      // B*F1*C*TnP = 16,515,072
    float* xgT  = h   + (size_t)Bn * F1n * Cn * TnP;       // B*TnP*C    =  1,032,192
    float* ax   = xgT + (size_t)Bn * TnP * Cn;             // B*C*TnP    =  1,032,192
    float* h2   = ax  + (size_t)Bn * Cn * TnP;             // B*C2*T     =    512,000
    float* s    = h2  + (size_t)Bn * C2n * Tn;             // B*C2       =        512
    float* att  = s   + (size_t)Bn * C2n;                  // B*C2       =        512
    float* hp   = att + (size_t)Bn * C2n;                  // B*C2*62    =     31,744
    float* xpad = hp  + (size_t)Bn * C2n * TPn;            // B*C*TPAD   =  1,097,728
    float* bnA  = xpad + (size_t)Bn * Cn * TPAD;           // F1         =         16
    float* bnC  = bnA + F1n;                               // F1         =         16

    // k0a: pad x -> x_pad; B*C*TPAD = 1,097,728 elems = 4288 blocks x 256
    k0_pad<<<(Bn * Cn * TPAD) / 256, 256, 0, stream>>>(x, xpad);
    // k0b: fold conv1 bias + BN1
    k0_prep<<<1, 32, 0, stream>>>(conv1b, g1, b1, m1, v1, bnA, bnC);
    // k1: 16*64*63 = 64512 waves; 8 waves/block -> 8064 blocks
    k1_conv1_bn_elu<<<8064, 256, 0, stream>>>(xpad, conv1w, bnA, bnC, h, xgT);
    // k2: 16*4*63 = 4032 waves -> 504 blocks
    k2_gcn<<<504, 256, 0, stream>>>(a_hat, xgT, ax);
    // k3: one block per (b, j)
    k3_dwexp<<<Bn * C2n, 256, 0, stream>>>(h, ax, dw_w, dw_b, gcn_w, gcn_b,
                                           g2, b2, m2, v2, h2, s);
    k4_att<<<1, Bn * C2n, 0, stream>>>(s, ca_w1, ca_b1, ca_w2, ca_b2, att);
    k5_spatial<<<Bn, 256, 0, stream>>>(h2, att, sa_w, sa_g, sa_b, sa_m, sa_v, hp);
    k6_sep<<<1, Bn * C2n, 0, stream>>>(hp, sep_w, sep_b, g3, b3, m3, v3, out);
}
